// LocalInferenceModeling_69054484185413
// MI455X (gfx1250) — matrix operand order repaired and checked by reading, not compiled, not run on hardware
//
#include <hip/hip_runtime.h>

#define NB 32
#define LL 512
#define DD 1024
#define LP 40   // LDS row pitch in bf16 elements (80B: 16B-aligned, conflict-free for b128 reads)

typedef __bf16  v8bf  __attribute__((ext_vector_type(8)));
typedef __bf16  v16bf __attribute__((ext_vector_type(16)));
typedef float   v8f   __attribute__((ext_vector_type(8)));
typedef int     v4i_  __attribute__((vector_size(16)));   // matches builtin param type

__device__ __forceinline__ unsigned short f2bf(float f){
  unsigned int u = __float_as_uint(f);
  u += 0x7fffu + ((u >> 16) & 1u);          // round-to-nearest-even
  return (unsigned short)(u >> 16);
}

__device__ __forceinline__ v16bf ld16(const unsigned short* p0, const unsigned short* p1){
  v8bf lo = *(const v8bf*)p0;
  v8bf hi = *(const v8bf*)p1;
  return __builtin_shufflevector(lo, hi, 0,1,2,3,4,5,6,7,8,9,10,11,12,13,14,15);
}

// ---- CDNA5 async global->LDS (ASYNCcnt) with compile-safe fallback ----
#if __has_builtin(__builtin_amdgcn_global_load_async_to_lds_b128)
#define HAVE_ASYNC_LDS 1
__device__ __forceinline__ void async_cp16(const unsigned short* g, unsigned short* l){
  __builtin_amdgcn_global_load_async_to_lds_b128(
      (__attribute__((address_space(1))) v4i_*)(void*)g,
      (__attribute__((address_space(3))) v4i_*)(void*)l, 0, 0);
}
__device__ __forceinline__ void wait_async_le2(){
#if __has_builtin(__builtin_amdgcn_s_wait_asynccnt)
  __builtin_amdgcn_s_wait_asynccnt(2);
#else
  asm volatile("s_wait_asynccnt 0x2" ::: "memory");
#endif
}
__device__ __forceinline__ void wait_async_0(){
#if __has_builtin(__builtin_amdgcn_s_wait_asynccnt)
  __builtin_amdgcn_s_wait_asynccnt(0);
#else
  asm volatile("s_wait_asynccnt 0x0" ::: "memory");
#endif
}
#else
#define HAVE_ASYNC_LDS 0
#endif

// ---------------- fp32 -> bf16 convert ----------------
__global__ __launch_bounds__(256) void cvt_bf16_kernel(const float* __restrict__ src,
                                                       unsigned short* __restrict__ dst, int n4){
  int i = blockIdx.x*256 + threadIdx.x;
  if (i >= n4) return;
  float4 f = ((const float4*)src)[i];
  unsigned int lo = (unsigned)f2bf(f.x) | ((unsigned)f2bf(f.y) << 16);
  unsigned int hi = (unsigned)f2bf(f.z) | ((unsigned)f2bf(f.w) << 16);
  ((uint2*)dst)[i] = make_uint2(lo, hi);
}

// ---------------- sim = P x H^T  (bf16 WMMA, fp32 accum, async double-buffer) ----------------
__global__ __launch_bounds__(256) void sim_kernel(const unsigned short* __restrict__ Pbf,
                                                  const unsigned short* __restrict__ Hbf,
                                                  float* __restrict__ sim){
  __shared__ unsigned short As[2][64*LP];
  __shared__ unsigned short Bs[2][64*LP];
  const int b   = blockIdx.z;
  const int p0  = blockIdx.x * 64;
  const int h0  = blockIdx.y * 64;
  const int tid = threadIdx.x;
  const int lane = tid & 31, w = tid >> 5;
  const int half = lane >> 4, ln = lane & 15;
  const int mrow = (w & 3) * 16;
  const int ncol = (w >> 2) * 32;

  const int lr = tid >> 2;        // 0..63 tile row
  const int lc = (tid & 3) * 8;   // bf16 chunk offset within 32-wide K tile
  const unsigned short* Ag = Pbf + ((size_t)b*LL + p0) * DD + (size_t)lr*DD + lc;
  const unsigned short* Bg = Hbf + ((size_t)b*LL + h0) * DD + (size_t)lr*DD + lc;
  const int lofs = lr*LP + lc;

  v8f c0 = {0,0,0,0,0,0,0,0};
  v8f c1 = {0,0,0,0,0,0,0,0};

  const int NT = DD / 32;

#if HAVE_ASYNC_LDS
  // prologue: tile 0 in flight
  async_cp16(Ag, &As[0][lofs]);
  async_cp16(Bg, &Bs[0][lofs]);
  #pragma unroll 2
  for (int i = 0; i < NT; ++i){
    const int cur = i & 1;
    if (i + 1 < NT){
      async_cp16(Ag + (i+1)*32, &As[1-cur][lofs]);   // overlap next tile with compute
      async_cp16(Bg + (i+1)*32, &Bs[1-cur][lofs]);
      wait_async_le2();                               // tile i complete (in-order), tile i+1 in flight
    } else {
      wait_async_0();
    }
    __syncthreads();
    v16bf a  = ld16(&As[cur][(mrow+ln)*LP + 16*half],    &As[cur][(mrow+ln)*LP + 16*half + 8]);
    v16bf b0 = ld16(&Bs[cur][(ncol+ln)*LP + 16*half],    &Bs[cur][(ncol+ln)*LP + 16*half + 8]);
    v16bf b1 = ld16(&Bs[cur][(ncol+16+ln)*LP + 16*half], &Bs[cur][(ncol+16+ln)*LP + 16*half + 8]);
    c0 = __builtin_amdgcn_wmma_f32_16x16x32_bf16(false, a, false, b0, (short)0, c0, false, false);
    c1 = __builtin_amdgcn_wmma_f32_16x16x32_bf16(false, a, false, b1, (short)0, c1, false, false);
    __syncthreads();
  }
#else
  for (int i = 0; i < NT; ++i){
    *(uint4*)&As[0][lofs] = *(const uint4*)(Ag + i*32);
    *(uint4*)&Bs[0][lofs] = *(const uint4*)(Bg + i*32);
    __builtin_prefetch(Ag + (i+1)*32, 0, 3);
    __builtin_prefetch(Bg + (i+1)*32, 0, 3);
    __syncthreads();
    v16bf a  = ld16(&As[0][(mrow+ln)*LP + 16*half],    &As[0][(mrow+ln)*LP + 16*half + 8]);
    v16bf b0 = ld16(&Bs[0][(ncol+ln)*LP + 16*half],    &Bs[0][(ncol+ln)*LP + 16*half + 8]);
    v16bf b1 = ld16(&Bs[0][(ncol+16+ln)*LP + 16*half], &Bs[0][(ncol+16+ln)*LP + 16*half + 8]);
    c0 = __builtin_amdgcn_wmma_f32_16x16x32_bf16(false, a, false, b0, (short)0, c0, false, false);
    c1 = __builtin_amdgcn_wmma_f32_16x16x32_bf16(false, a, false, b1, (short)0, c1, false, false);
    __syncthreads();
  }
#endif

  float* sb = sim + (size_t)b*LL*LL;
  #pragma unroll
  for (int v = 0; v < 8; ++v){
    int p = p0 + mrow + v + 8*half;
    sb[(size_t)p*LL + h0 + ncol + ln]      = c0[v];
    sb[(size_t)p*LL + h0 + ncol + 16 + ln] = c1[v];
  }
}

// ---------------- row softmax: W1[b,p,h], kv mask = hypothesis_mask ----------------
__global__ __launch_bounds__(256) void row_softmax_kernel(const float* __restrict__ sim,
                                                          const float* __restrict__ kvmask,
                                                          float* __restrict__ W){
  __shared__ float r1[256];
  __shared__ float r2[256];
  const int b = blockIdx.y, p = blockIdx.x, tid = threadIdx.x;
  const float* row = sim + ((size_t)b*LL + p)*LL;
  const float* msk = kvmask + (size_t)b*LL;
  float m0 = msk[tid], m1 = msk[tid+256];
  float s0 = row[tid]*m0, s1 = row[tid+256]*m1;
  float mx = fmaxf(s0, s1);
  r1[tid] = mx; __syncthreads();
  for (int s = 128; s > 0; s >>= 1){ if (tid < s) r1[tid] = fmaxf(r1[tid], r1[tid+s]); __syncthreads(); }
  mx = r1[0]; __syncthreads();
  float e0 = __expf(s0-mx), e1 = __expf(s1-mx);
  r1[tid] = e0 + e1;
  r2[tid] = e0*m0 + e1*m1;
  __syncthreads();
  for (int s = 128; s > 0; s >>= 1){ if (tid < s){ r1[tid]+=r1[tid+s]; r2[tid]+=r2[tid+s]; } __syncthreads(); }
  float inv = 1.0f / (r2[0] + 1e-13f*r1[0]);
  float* wr = W + ((size_t)b*LL + p)*LL;
  wr[tid]     = e0*m0*inv;
  wr[tid+256] = e1*m1*inv;
}

// ------- column softmax in place: sim[b][p][h] := W2(h,p)  (stores W2^T) -------
__global__ __launch_bounds__(256) void col_softmax_kernel(float* __restrict__ sim,
                                                          const float* __restrict__ kvmask){
  __shared__ float r1[256];
  __shared__ float r2[256];
  const int b = blockIdx.y, tid = threadIdx.x;
  const int col = blockIdx.x*64 + (tid & 63);
  const int seg = tid >> 6;                 // 4 segments of 128 rows
  float* sb = sim + (size_t)b*LL*LL;
  const float* msk = kvmask + (size_t)b*LL;
  float mx = -3.0e38f;
  for (int r = seg*128; r < seg*128+128; ++r)
    mx = fmaxf(mx, sb[(size_t)r*LL + col] * msk[r]);
  r1[tid] = mx; __syncthreads();
  const int ci = tid & 63;
  mx = fmaxf(fmaxf(r1[ci], r1[64+ci]), fmaxf(r1[128+ci], r1[192+ci]));
  __syncthreads();
  float se = 0.f, sem = 0.f;
  for (int r = seg*128; r < seg*128+128; ++r){
    float m = msk[r];
    float e = __expf(sb[(size_t)r*LL + col]*m - mx);
    se += e; sem += e*m;
  }
  r1[tid] = se; r2[tid] = sem; __syncthreads();
  se  = r1[ci] + r1[64+ci] + r1[128+ci] + r1[192+ci];
  sem = r2[ci] + r2[64+ci] + r2[128+ci] + r2[192+ci];
  float inv = 1.0f / (sem + 1e-13f*se);
  for (int r = seg*128; r < seg*128+128; ++r){
    float m = msk[r];
    float e = __expf(sb[(size_t)r*LL + col]*m - mx);
    sb[(size_t)r*LL + col] = e*m*inv;
  }
}

// ------- attended_premises = W1 @ H, fused compose [P, AP, P-AP, P*AP] -------
__global__ __launch_bounds__(256) void ap_compose_kernel(const float* __restrict__ W1,
                                                         const unsigned short* __restrict__ Hbf,
                                                         const float* __restrict__ Pf,
                                                         const float* __restrict__ qmask,
                                                         float* __restrict__ out){
  __shared__ unsigned short As[64*LP];
  __shared__ unsigned short Bs[64*LP];
  const int b  = blockIdx.z;
  const int p0 = blockIdx.y * 64;
  const int d0 = blockIdx.x * 64;
  const int tid = threadIdx.x;
  const int lane = tid & 31, w = tid >> 5;
  const int half = lane >> 4, ln = lane & 15;
  const int mrow = (w & 3)*16, ncol = (w >> 2)*32;

  const float* Ab = W1 + ((size_t)b*LL + p0)*LL;
  const unsigned short* Hb = Hbf + (size_t)b*LL*DD;

  v8f c0 = {0,0,0,0,0,0,0,0};
  v8f c1 = {0,0,0,0,0,0,0,0};

  const int ar = tid >> 3;        // 0..31
  const int ac = (tid & 7)*4;     // f32 K chunk
  const int bk = tid >> 3;        // 0..31 (B k row)
  const int bc = (tid & 7)*8;     // n offset

  for (int k0 = 0; k0 < LL; k0 += 32){
    #pragma unroll
    for (int j = 0; j < 2; ++j){  // stage A: f32 -> bf16, row-major
      int m = ar + 32*j;
      float4 f = *(const float4*)(Ab + (size_t)m*LL + k0 + ac);
      *(unsigned int*)&As[m*LP + ac]     = (unsigned)f2bf(f.x) | ((unsigned)f2bf(f.y)<<16);
      *(unsigned int*)&As[m*LP + ac + 2] = (unsigned)f2bf(f.z) | ((unsigned)f2bf(f.w)<<16);
      __builtin_prefetch(Ab + (size_t)m*LL + k0 + 32 + ac, 0, 3);
    }
    { // stage B transposed: Bs[n][k] = Hbf[k0+k][d0+n]
      uint4 v = *(const uint4*)(Hb + (size_t)(k0+bk)*DD + d0 + bc);
      __builtin_prefetch(Hb + (size_t)(k0+32+bk)*DD + d0 + bc, 0, 3);
      unsigned short e[8] = {
        (unsigned short)(v.x&0xffffu),(unsigned short)(v.x>>16),
        (unsigned short)(v.y&0xffffu),(unsigned short)(v.y>>16),
        (unsigned short)(v.z&0xffffu),(unsigned short)(v.z>>16),
        (unsigned short)(v.w&0xffffu),(unsigned short)(v.w>>16)};
      #pragma unroll
      for (int j = 0; j < 8; ++j) Bs[(bc+j)*LP + bk] = e[j];
    }
    __syncthreads();
    v16bf a  = ld16(&As[(mrow+ln)*LP + 16*half],    &As[(mrow+ln)*LP + 16*half + 8]);
    v16bf b0 = ld16(&Bs[(ncol+ln)*LP + 16*half],    &Bs[(ncol+ln)*LP + 16*half + 8]);
    v16bf b1 = ld16(&Bs[(ncol+16+ln)*LP + 16*half], &Bs[(ncol+16+ln)*LP + 16*half + 8]);
    c0 = __builtin_amdgcn_wmma_f32_16x16x32_bf16(false, a, false, b0, (short)0, c0, false, false);
    c1 = __builtin_amdgcn_wmma_f32_16x16x32_bf16(false, a, false, b1, (short)0, c1, false, false);
    __syncthreads();
  }
  const float* xb = Pf + (size_t)b*LL*DD;
  const float* qm = qmask + (size_t)b*LL;
  float* ob = out + (size_t)b*LL*4096;
  #pragma unroll
  for (int v = 0; v < 8; ++v){
    int p = p0 + mrow + v + 8*half;
    float q = qm[p];
    int d = d0 + ncol + ln;
    size_t o = (size_t)p*4096 + d;
    float x0 = xb[(size_t)p*DD + d];
    float a0 = c0[v]*q;
    ob[o] = x0; ob[o+1024] = a0; ob[o+2048] = x0 - a0; ob[o+3072] = x0 * a0;
    float x1 = xb[(size_t)p*DD + d + 16];
    float a1 = c1[v]*q;
    ob[o+16] = x1; ob[o+16+1024] = a1; ob[o+16+2048] = x1 - a1; ob[o+16+3072] = x1 * a1;
  }
}

// ------- attended_hypotheses = W2 @ P (A stored transposed), fused compose -------
__global__ __launch_bounds__(256) void ah_compose_kernel(const float* __restrict__ W2T,
                                                         const unsigned short* __restrict__ Pbf,
                                                         const float* __restrict__ Hf,
                                                         const float* __restrict__ qmask,
                                                         float* __restrict__ out){
  __shared__ unsigned short As[64*LP];
  __shared__ unsigned short Bs[64*LP];
  const int b  = blockIdx.z;
  const int h0 = blockIdx.y * 64;
  const int d0 = blockIdx.x * 64;
  const int tid = threadIdx.x;
  const int lane = tid & 31, w = tid >> 5;
  const int half = lane >> 4, ln = lane & 15;
  const int mrow = (w & 3)*16, ncol = (w >> 2)*32;

  const float* Ab = W2T + (size_t)b*LL*LL;        // [p*512 + h] = W2(h,p)
  const unsigned short* Pb = Pbf + (size_t)b*LL*DD;

  v8f c0 = {0,0,0,0,0,0,0,0};
  v8f c1 = {0,0,0,0,0,0,0,0};

  const int kk = tid >> 4;        // 0..15 (A k half-row)
  const int mc = (tid & 15)*4;    // m chunk (4 h values)
  const int bk = tid >> 3;        // 0..31
  const int bc = (tid & 7)*8;

  for (int k0 = 0; k0 < LL; k0 += 32){
    #pragma unroll
    for (int j = 0; j < 2; ++j){  // stage A transposed: As[m=h][k=p], cvt f32->bf16
      int k = kk + 16*j;
      float4 f = *(const float4*)(Ab + (size_t)(k0+k)*LL + h0 + mc);
      __builtin_prefetch(Ab + (size_t)(k0+32+k)*LL + h0 + mc, 0, 3);
      As[(mc+0)*LP + k] = f2bf(f.x);
      As[(mc+1)*LP + k] = f2bf(f.y);
      As[(mc+2)*LP + k] = f2bf(f.z);
      As[(mc+3)*LP + k] = f2bf(f.w);
    }
    { // stage B transposed: Bs[n][k] = Pbf[k0+k][d0+n]
      uint4 v = *(const uint4*)(Pb + (size_t)(k0+bk)*DD + d0 + bc);
      __builtin_prefetch(Pb + (size_t)(k0+32+bk)*DD + d0 + bc, 0, 3);
      unsigned short e[8] = {
        (unsigned short)(v.x&0xffffu),(unsigned short)(v.x>>16),
        (unsigned short)(v.y&0xffffu),(unsigned short)(v.y>>16),
        (unsigned short)(v.z&0xffffu),(unsigned short)(v.z>>16),
        (unsigned short)(v.w&0xffffu),(unsigned short)(v.w>>16)};
      #pragma unroll
      for (int j = 0; j < 8; ++j) Bs[(bc+j)*LP + bk] = e[j];
    }
    __syncthreads();
    v16bf a  = ld16(&As[(mrow+ln)*LP + 16*half],    &As[(mrow+ln)*LP + 16*half + 8]);
    v16bf b0 = ld16(&Bs[(ncol+ln)*LP + 16*half],    &Bs[(ncol+ln)*LP + 16*half + 8]);
    v16bf b1 = ld16(&Bs[(ncol+16+ln)*LP + 16*half], &Bs[(ncol+16+ln)*LP + 16*half + 8]);
    c0 = __builtin_amdgcn_wmma_f32_16x16x32_bf16(false, a, false, b0, (short)0, c0, false, false);
    c1 = __builtin_amdgcn_wmma_f32_16x16x32_bf16(false, a, false, b1, (short)0, c1, false, false);
    __syncthreads();
  }
  const float* xb = Hf + (size_t)b*LL*DD;
  const float* qm = qmask + (size_t)b*LL;
  float* ob = out + (size_t)b*LL*4096;
  #pragma unroll
  for (int v = 0; v < 8; ++v){
    int h = h0 + mrow + v + 8*half;
    float q = qm[h];
    int d = d0 + ncol + ln;
    size_t o = (size_t)h*4096 + d;
    float x0 = xb[(size_t)h*DD + d];
    float a0 = c0[v]*q;
    ob[o] = x0; ob[o+1024] = a0; ob[o+2048] = x0 - a0; ob[o+3072] = x0 * a0;
    float x1 = xb[(size_t)h*DD + d + 16];
    float a1 = c1[v]*q;
    ob[o+16] = x1; ob[o+16+1024] = a1; ob[o+16+2048] = x1 - a1; ob[o+16+3072] = x1 * a1;
  }
}

extern "C" void kernel_launch(void* const* d_in, const int* in_sizes, int n_in,
                              void* d_out, int out_size, void* d_ws, size_t ws_size,
                              hipStream_t stream){
  const float* P  = (const float*)d_in[0];
  const float* H  = (const float*)d_in[1];
  const float* pm = (const float*)d_in[2];
  const float* hm = (const float*)d_in[3];
  float* out = (float*)d_out;

  const size_t NPH  = (size_t)NB*LL*DD;   // 16,777,216
  const size_t NSIM = (size_t)NB*LL*LL;   //  8,388,608
  unsigned short* Pbf = (unsigned short*)d_ws;
  unsigned short* Hbf = Pbf + NPH;
  float* sim = (float*)(Hbf + NPH);
  float* W1  = sim + NSIM;

  const int n4 = (int)(NPH/4);
  cvt_bf16_kernel<<<(n4+255)/256, 256, 0, stream>>>(P, Pbf, n4);
  cvt_bf16_kernel<<<(n4+255)/256, 256, 0, stream>>>(H, Hbf, n4);
  sim_kernel<<<dim3(LL/64, LL/64, NB), 256, 0, stream>>>(Pbf, Hbf, sim);
  row_softmax_kernel<<<dim3(LL, NB), 256, 0, stream>>>(sim, hm, W1);
  col_softmax_kernel<<<dim3(LL/64, NB), 256, 0, stream>>>(sim, pm);
  ap_compose_kernel<<<dim3(DD/64, LL/64, NB), 256, 0, stream>>>(W1, Hbf, P, pm, out);
  ah_compose_kernel<<<dim3(DD/64, LL/64, NB), 256, 0, stream>>>(sim, Pbf, H, hm,
                                                                out + (size_t)NB*LL*4096);
}
